// TreeConvolution_13091060318997
// MI455X (gfx1250) — matrix operand order repaired and checked by reading, not compiled, not run on hardware
//
#include <hip/hip_runtime.h>

// ---- CDNA5 (gfx1250) wave32 WMMA types ----
typedef __attribute__((ext_vector_type(16))) __bf16 v16bf;
typedef __attribute__((ext_vector_type(8)))  __bf16 v8bf;
typedef __attribute__((ext_vector_type(4)))  __bf16 v4bf;
typedef __attribute__((ext_vector_type(2)))  __bf16 v2bf;
typedef __attribute__((ext_vector_type(8)))  float  v8f;
typedef __attribute__((ext_vector_type(4)))  float  f4v;

union Frag16 { v16bf v; v8bf h[2]; };

constexpr int BN = 32;    // batch
constexpr int CC = 64;    // channels
constexpr int LL = 2048;  // length
constexpr int OC = 128;   // out channels
constexpr int KK = 192;   // 3*CC

// ---------------- f32 -> bf16 conversion (4 elems/thread) ----------------
__global__ __launch_bounds__(256) void cvt4_kernel(const float* __restrict__ src,
                                                   __bf16* __restrict__ dst, int n4) {
  int i = blockIdx.x * 256 + threadIdx.x;
  if (i < n4) {
    const f4v f = *(const f4v*)(src + (size_t)i * 4);
    v4bf o;
    o[0] = (__bf16)f[0]; o[1] = (__bf16)f[1]; o[2] = (__bf16)f[2]; o[3] = (__bf16)f[3];
    *(v4bf*)(dst + (size_t)i * 4) = o;
  }
}

// ---------------- y = A(64xL,bf16) @ M(LxL,f32) -> bf16 ----------------
// 256 thr (8 waves), tile 64x128, KT=32. Double-buffered LDS, one barrier/iter,
// global prefetch in regs overlaps WMMAs; m streamed with b128 NT loads.
__global__ __launch_bounds__(256) void prop_gemm(const __bf16* __restrict__ Abf,
                                                 const float* __restrict__ Mf,
                                                 __bf16* __restrict__ Ybf) {
  constexpr int KT = 32, NT = 128, LDK = KT + 8;      // rows 80B (16B aligned)
  __shared__ __align__(16) __bf16 Alds[2][CC][LDK];   // [buf][row][k]
  __shared__ __align__(16) __bf16 BldsT[2][NT][LDK];  // [buf][col][k] (transposed)

  const int n    = blockIdx.y;
  const int j0   = blockIdx.x * NT;
  const int tid  = threadIdx.x;
  const int lane = tid & 31;
  const int wave = tid >> 5;       // 0..7
  const int wm   = wave & 3;       // M block (rows wm*16..)
  const int wj   = wave >> 2;      // N half (cols wj*64..)
  const int fr   = lane & 15;      // A: row-in-16 / B: col-in-16 / C/D: N
  const int akb  = (lane >> 4) * 8;   // A frag K base (ISA 16-bit A layout)
  const int bkb  = (lane >> 4) * 8;   // B frag K base (ISA 16-bit B layout)

  const __bf16* An = Abf + (size_t)n * CC * LL;
  const float*  Mn = Mf  + (size_t)n * LL * LL;

  v8f acc[4];
#pragma unroll
  for (int t = 0; t < 4; ++t) { v8f z = {}; acc[t] = z; }

  const int ar  = tid >> 2, ac = (tid & 3) * 8;          // A loader: row, 8-half chunk
  const int c4  = (tid & 31) * 4, kk4 = (tid >> 5) * 4;  // B loader: 4 cols x 4 k-rows

  // register prefetch buffers (global -> regs -> LDS pipeline)
  v8bf aReg;
  f4v  mReg[4];

  auto gload = [&](int k0) {
    aReg = *(const v8bf*)(An + (size_t)ar * LL + k0 + ac);
#pragma unroll
    for (int r = 0; r < 4; ++r) {
      const f4v* gp = (const f4v*)(Mn + (size_t)(k0 + kk4 + r) * LL + j0 + c4);
      mReg[r] = __builtin_nontemporal_load(gp);   // stream m: b128, non-temporal
    }
  };
  auto commit = [&](int buf) {
    *(v8bf*)&Alds[buf][ar][ac] = aReg;
#pragma unroll
    for (int c = 0; c < 4; ++c) {
      v4bf p;
#pragma unroll
      for (int r = 0; r < 4; ++r) p[r] = (__bf16)mReg[r][c];
      *(v4bf*)&BldsT[buf][c4 + c][kk4] = p;       // 8B transposed commit
    }
  };

  gload(0);
  commit(0);
  __syncthreads();
  gload(KT);

  for (int k0 = 0; k0 < LL; k0 += KT) {
    const int buf = (k0 >> 5) & 1;

    // gather all fragments, then back-to-back WMMAs (one dscnt wait amortized)
    Frag16 af;
    af.h[0] = *(const v8bf*)&Alds[buf][wm * 16 + fr][akb];
    af.h[1] = *(const v8bf*)&Alds[buf][wm * 16 + fr][akb + 16];
    Frag16 bq[4];
#pragma unroll
    for (int t = 0; t < 4; ++t) {
      const int col = wj * 64 + t * 16 + fr;
      bq[t].h[0] = *(const v8bf*)&BldsT[buf][col][bkb];
      bq[t].h[1] = *(const v8bf*)&BldsT[buf][col][bkb + 16];
    }
#pragma unroll
    for (int t = 0; t < 4; ++t)
      acc[t] = __builtin_amdgcn_wmma_f32_16x16x32_bf16(
          false, af.v, false, bq[t].v, (short)0, acc[t], false, false);

    if (k0 + KT < LL) {
      commit(buf ^ 1);          // tile k0+KT into other buffer
      __syncthreads();          // single barrier per iteration
      if (k0 + 2 * KT < LL) gload(k0 + 2 * KT);
    }
  }

  // C/D layout: VGPR r -> M = r + 8*(lane>=16); N = lane%16
  __bf16* Yn = Ybf + (size_t)n * CC * LL;
  const int rbase = wm * 16 + ((lane >> 4) << 3);
  const int jj    = j0 + wj * 64 + fr;
#pragma unroll
  for (int t = 0; t < 4; ++t)
#pragma unroll
    for (int r = 0; r < 8; ++r)
      Yn[(size_t)(rbase + r) * LL + jj + t * 16] = (__bf16)acc[t][r];
}

// ---------------- out = W(128x192,bf16) @ cat(192xL,bf16) + b -> f32 ----------------
// 256 thr (8 waves), tile 128x64, K=192 in 6 steps of 32. Same pipeline shape.
__global__ __launch_bounds__(256) void proj_gemm(const __bf16* __restrict__ Wbf,
                                                 const __bf16* __restrict__ X0,
                                                 const __bf16* __restrict__ X1,
                                                 const __bf16* __restrict__ X2,
                                                 const float* __restrict__ bias,
                                                 float* __restrict__ Out) {
  constexpr int KT = 32, NT = 64, LDK = KT + 8;
  __shared__ __align__(16) __bf16 Alds[2][OC][LDK];
  __shared__ __align__(16) __bf16 BldsT[2][NT][LDK];

  const int n    = blockIdx.y;
  const int j0   = blockIdx.x * NT;
  const int tid  = threadIdx.x;
  const int lane = tid & 31;
  const int wave = tid >> 5;   // = M block 0..7
  const int fr   = lane & 15;
  const int akb  = (lane >> 4) * 8;
  const int bkb  = (lane >> 4) * 8;

  const __bf16* srcs[3] = {X0 + (size_t)n * CC * LL,
                           X1 + (size_t)n * CC * LL,
                           X2 + (size_t)n * CC * LL};

  v8f acc[4];
#pragma unroll
  for (int t = 0; t < 4; ++t) { v8f z = {}; acc[t] = z; }

  const int ar = tid >> 1, ac = (tid & 1) * 16;          // A loader: row, 16-half chunk
  const int c4 = (tid & 15) * 4, kk2 = (tid >> 4) * 2;   // B loader: 4 cols x 2 k-rows

  v8bf wReg0, wReg1;
  v4bf bReg[2];
  auto gload = [&](int k0) {
    wReg0 = *(const v8bf*)(Wbf + (size_t)ar * KK + k0 + ac);
    wReg1 = *(const v8bf*)(Wbf + (size_t)ar * KK + k0 + ac + 8);
    const __bf16* Bsrc = srcs[k0 >> 6];
    const int kl = k0 & 63;
#pragma unroll
    for (int r = 0; r < 2; ++r)
      bReg[r] = *(const v4bf*)(Bsrc + (size_t)(kl + kk2 + r) * LL + j0 + c4);
  };
  auto commit = [&](int buf) {
    *(v8bf*)&Alds[buf][ar][ac]     = wReg0;
    *(v8bf*)&Alds[buf][ar][ac + 8] = wReg1;
#pragma unroll
    for (int c = 0; c < 4; ++c) {
      v2bf p; p[0] = bReg[0][c]; p[1] = bReg[1][c];
      *(v2bf*)&BldsT[buf][c4 + c][kk2] = p;
    }
  };

  gload(0);
  commit(0);
  __syncthreads();
  gload(KT);

  for (int k0 = 0; k0 < KK; k0 += KT) {
    const int buf = (k0 >> 5) & 1;

    Frag16 af;
    af.h[0] = *(const v8bf*)&Alds[buf][wave * 16 + fr][akb];
    af.h[1] = *(const v8bf*)&Alds[buf][wave * 16 + fr][akb + 16];
    Frag16 bq[4];
#pragma unroll
    for (int t = 0; t < 4; ++t) {
      const int col = t * 16 + fr;
      bq[t].h[0] = *(const v8bf*)&BldsT[buf][col][bkb];
      bq[t].h[1] = *(const v8bf*)&BldsT[buf][col][bkb + 16];
    }
#pragma unroll
    for (int t = 0; t < 4; ++t)
      acc[t] = __builtin_amdgcn_wmma_f32_16x16x32_bf16(
          false, af.v, false, bq[t].v, (short)0, acc[t], false, false);

    if (k0 + KT < KK) {
      commit(buf ^ 1);
      __syncthreads();
      if (k0 + 2 * KT < KK) gload(k0 + 2 * KT);
    }
  }

  const int rbase = wave * 16 + ((lane >> 4) << 3);
  float bv[8];
#pragma unroll
  for (int r = 0; r < 8; ++r) bv[r] = bias[rbase + r];
  float* On = Out + (size_t)n * OC * LL;
#pragma unroll
  for (int t = 0; t < 4; ++t)
#pragma unroll
    for (int r = 0; r < 8; ++r)
      On[(size_t)(rbase + r) * LL + j0 + t * 16 + fr] = acc[t][r] + bv[r];
}

extern "C" void kernel_launch(void* const* d_in, const int* in_sizes, int n_in,
                              void* d_out, int out_size, void* d_ws, size_t ws_size,
                              hipStream_t stream) {
  const float* inp = (const float*)d_in[0];  // (32,64,2048)
  const float* m   = (const float*)d_in[1];  // (32,2048,2048)
  const float* W   = (const float*)d_in[2];  // (128,192)
  const float* b   = (const float*)d_in[3];  // (128,)
  float* out = (float*)d_out;                // (32,128,2048)

  char* ws = (char*)d_ws;
  __bf16* inp_bf = (__bf16*)(ws);
  __bf16* y1_bf  = (__bf16*)(ws + (size_t)8  * 1024 * 1024);
  __bf16* y2_bf  = (__bf16*)(ws + (size_t)16 * 1024 * 1024);
  __bf16* W_bf   = (__bf16*)(ws + (size_t)24 * 1024 * 1024);

  const int n_inp4 = BN * CC * LL / 4;   // 1,048,576
  const int n_w4   = OC * KK / 4;        // 6,144
  cvt4_kernel<<<(n_inp4 + 255) / 256, 256, 0, stream>>>(inp, inp_bf, n_inp4);
  cvt4_kernel<<<(n_w4 + 255) / 256, 256, 0, stream>>>(W, W_bf, n_w4);

  dim3 gp(LL / 128, BN);  // (16, 32)
  prop_gemm<<<gp, 256, 0, stream>>>(inp_bf, m, y1_bf);
  prop_gemm<<<gp, 256, 0, stream>>>(y1_bf, m, y2_bf);

  dim3 go(LL / 64, BN);   // (32, 32)
  proj_gemm<<<go, 256, 0, stream>>>(W_bf, inp_bf, y1_bf, y2_bf, b, out);
}
